// Grok3pAttention_72816875536952
// MI455X (gfx1250) — matrix operand order repaired and checked by reading, not compiled, not run on hardware
//
#include <hip/hip_runtime.h>

// ---------------------------------------------------------------------------
// CDNA5 (gfx1250, wave32) attention layer.
// GEMM math: V_WMMA_F32_16X16X32_BF16 (fp32 accumulate).
// Data movement: GLOBAL_LOAD_ASYNC_TO_LDS_B128 double-buffered tile staging
// (ASYNCcnt), fragments served from LDS at conflict-free padded pitches.
// Workspace layout (117,440,512 bytes total):
//   [0,16MB)      hs_bf16    [4096,2048]
//   [16,41.9MB)   Wqkv_bf16  [6144,2048]
//   [41.9,50.3MB) Wout_bf16  [2048,2048]
//   [50.3MB..)    q_bf16, k_bf16 [B,H,S,D]; v_t_bf16 [B,H,D,S]; ctx_bf16 [4096,2048]
// ---------------------------------------------------------------------------

typedef __bf16 bf16_t;
typedef bf16_t bf16x8  __attribute__((ext_vector_type(8)));
typedef bf16_t bf16x16 __attribute__((ext_vector_type(16)));
typedef float  f32x8   __attribute__((ext_vector_type(8)));

union Frag16 { bf16x16 v; bf16x8 h[2]; };

__device__ __forceinline__ bf16_t to_bf16(float f) {
  union { unsigned u; float f; } in; in.f = f;
  unsigned r = (in.u + 0x7FFFu + ((in.u >> 16) & 1u)) >> 16;   // RNE
  union { unsigned short s; bf16_t b; } out; out.s = (unsigned short)r;
  return out.b;
}

__device__ __forceinline__ f32x8 wmma_bf16(bf16x16 a, bf16x16 b, f32x8 c) {
  return __builtin_amdgcn_wmma_f32_16x16x32_bf16(false, a, false, b,
                                                 (short)0, c, false, false);
}

// ---- CDNA5 async copy: 16 bytes/lane global -> LDS, tracked by ASYNCcnt ----
__device__ __forceinline__ void async_copy16(const bf16_t* g, unsigned lds_off) {
  asm volatile("global_load_async_to_lds_b128 %0, %1, off"
               :: "v"(lds_off), "v"((unsigned long long)(size_t)g)
               : "memory");
}
__device__ __forceinline__ void wait_async0() {
  asm volatile("s_wait_asynccnt 0" ::: "memory");
}
// Low 32 bits of a generic LDS pointer == LDS byte offset (ISA 10.2 aperture).
__device__ __forceinline__ unsigned lds_off_of(const void* p) {
  return (unsigned)(size_t)p;
}

// A fragment: 16(M) x 32(K); source row-major, leading dim ld (elements).
// Layout used here (contiguous k): lanes 0-15 hold M=lane, K=[0,16); lanes 16-31 K=[16,32).
__device__ __forceinline__ bf16x16 load_frag_a(const bf16_t* p, int ld, int lane) {
  const int lrow = lane & 15, hi = lane >> 4;
  const bf16_t* q = p + (size_t)lrow * ld + 16 * hi;
  Frag16 f;
  f.h[0] = *(const bf16x8*)(q);
  f.h[1] = *(const bf16x8*)(q + 8);
  return f.v;
}

// B fragment: 32(K) x 16(N); source stored N-major with K contiguous (ld elems/row).
// Layout used here (contiguous k): lanes 0-15 hold N=lane, K=[0,16); lanes 16-31 K=[16,32).
__device__ __forceinline__ bf16x16 load_frag_b(const bf16_t* p, int ld, int lane) {
  const int lrow = lane & 15, hi = lane >> 4;
  const bf16_t* q = p + (size_t)lrow * ld + 16 * hi;
  return *(const bf16x16*)(q);
}

// ---------------------------------------------------------------------------
// fp32 -> bf16 conversion
// ---------------------------------------------------------------------------
__global__ void cvt_f32_bf16_kernel(const float* __restrict__ in,
                                    bf16_t* __restrict__ out, int n) {
  int i = blockIdx.x * blockDim.x + threadIdx.x;
  if (i < n) out[i] = to_bf16(in[i]);
}

// ---------------------------------------------------------------------------
// GEMM: C[M,N] = A[M,K] @ W[N,K]^T + bias.
// mode 0: write fp32 row-major to Cout.
// mode 1: (QKV) scatter bf16: q,k -> [B,H,S,D]; v -> [B,H,D,S] (transposed).
// Block = 256 threads = 8 waves (2x4), block tile 128x256, k-step 32.
// Tiles staged in LDS (double buffered) via async copies:
//   A tile 128 rows x 64B @ pitch 80B  = 10240 B
//   B tile 256 rows x 64B @ pitch 80B  = 20480 B   -> 30720 B per buffer
// ---------------------------------------------------------------------------
#define G_APITCH 40            // elements (80 B)
#define G_BUFSZ  30720u

__device__ __forceinline__ void gemm_issue_tile(
    const bf16_t* __restrict__ A, const bf16_t* __restrict__ W,
    int m0b, int n0b, int K, int k0, unsigned smem_base, int buf, int tid) {
  const unsigned abase = smem_base + (unsigned)buf * G_BUFSZ;
  const unsigned bbase = abase + 10240u;
#pragma unroll
  for (int u = 0; u < 2; ++u) {                       // A: 512 16B chunks
    const int idx = tid + 256 * u;
    const int row = idx >> 2, c = idx & 3;
    async_copy16(A + (size_t)(m0b + row) * K + k0 + c * 8,
                 abase + (unsigned)row * 80u + (unsigned)c * 16u);
  }
#pragma unroll
  for (int u = 0; u < 4; ++u) {                       // B: 1024 16B chunks
    const int idx = tid + 256 * u;
    const int row = idx >> 2, c = idx & 3;
    async_copy16(W + (size_t)(n0b + row) * K + k0 + c * 8,
                 bbase + (unsigned)row * 80u + (unsigned)c * 16u);
  }
}

__global__ __launch_bounds__(256) void gemm_bf16_kernel(
    const bf16_t* __restrict__ A, const bf16_t* __restrict__ W,
    const float* __restrict__ bias, float* __restrict__ Cout,
    bf16_t* __restrict__ qo, bf16_t* __restrict__ ko, bf16_t* __restrict__ vt,
    int M, int N, int K, int mode) {
  __shared__ __align__(16) char gsm[2 * G_BUFSZ];
  const int tid = threadIdx.x;
  const int lane = tid & 31;
  const int wave = tid >> 5;
  const int lrow = lane & 15, hi = lane >> 4;
  const int waveM = wave >> 2, waveN = wave & 3;
  const int m0b = blockIdx.y * 128;
  const int n0b = blockIdx.x * 256;
  const int m0 = m0b + waveM * 64;
  const int n0 = n0b + waveN * 64;
  const unsigned smem_base = lds_off_of(gsm);

  const f32x8 zero = {};
  f32x8 acc[4][4];
#pragma unroll
  for (int i = 0; i < 4; ++i)
#pragma unroll
    for (int j = 0; j < 4; ++j) acc[i][j] = zero;

  const int nk = K >> 5;
  gemm_issue_tile(A, W, m0b, n0b, K, 0, smem_base, 0, tid);

  for (int kt = 0; kt < nk; ++kt) {
    const int cur = kt & 1;
    wait_async0();
    __syncthreads();
    if (kt + 1 < nk)
      gemm_issue_tile(A, W, m0b, n0b, K, (kt + 1) << 5, smem_base, cur ^ 1, tid);

    const bf16_t* AL = (const bf16_t*)(gsm + cur * G_BUFSZ);
    const bf16_t* BL = (const bf16_t*)(gsm + cur * G_BUFSZ + 10240);
    bf16x16 af[4];
#pragma unroll
    for (int i = 0; i < 4; ++i)
      af[i] = load_frag_a(AL + (waveM * 64 + 16 * i) * G_APITCH, G_APITCH, lane);
#pragma unroll
    for (int j = 0; j < 4; ++j) {
      bf16x16 bfm = load_frag_b(BL + (waveN * 64 + 16 * j) * G_APITCH, G_APITCH, lane);
#pragma unroll
      for (int i = 0; i < 4; ++i) acc[i][j] = wmma_bf16(af[i], bfm, acc[i][j]);
    }
  }

#pragma unroll
  for (int j = 0; j < 4; ++j) {
    const int n = n0 + 16 * j + lrow;
    const float bi = bias[n];
    if (mode == 0) {
#pragma unroll
      for (int i = 0; i < 4; ++i)
#pragma unroll
        for (int r = 0; r < 8; ++r) {
          const int m = m0 + 16 * i + r + 8 * hi;
          Cout[(size_t)m * N + n] = acc[i][j][r] + bi;
        }
    } else {
      const int t = n >> 11;           // 0=q, 1=k, 2=v
      const int h = (n >> 7) & 15;
      const int d = n & 127;
#pragma unroll
      for (int i = 0; i < 4; ++i)
#pragma unroll
        for (int r = 0; r < 8; ++r) {
          const int m = m0 + 16 * i + r + 8 * hi;     // global token index
          const int bb = m >> 11, s = m & 2047;
          const bf16_t val = to_bf16(acc[i][j][r] + bi);
          const size_t bh = (size_t)(bb * 16 + h);
          if (t == 0)      qo[(bh * 2048 + s) * 128 + d] = val;
          else if (t == 1) ko[(bh * 2048 + s) * 128 + d] = val;
          else             vt[(bh * 128 + d) * 2048 + s] = val;
        }
    }
  }
}

// ---------------------------------------------------------------------------
// In-place RoPE on q,k stored bf16 [B*H, S, 128]. One thread per (bh, s, pair).
// ---------------------------------------------------------------------------
__global__ void rope_kernel(bf16_t* __restrict__ q, bf16_t* __restrict__ k) {
  const int idx = blockIdx.x * blockDim.x + threadIdx.x;   // 2^22 total
  const int i  = idx & 63;
  const int s  = (idx >> 6) & 2047;
  const int bh = idx >> 17;
  const float inv = __expf(-(2.0f * (float)i / 128.0f) * 9.210340371976184f);
  const float ang = (float)s * inv;
  float sn, c;
  __sincosf(ang, &sn, &c);
  const size_t base = ((size_t)bh * 2048 + s) * 128 + 2 * i;
  const float qe = (float)q[base], qo = (float)q[base + 1];
  q[base]     = to_bf16(qe * c - qo * sn);
  q[base + 1] = to_bf16(qe * sn + qo * c);
  const float ke = (float)k[base], ko = (float)k[base + 1];
  k[base]     = to_bf16(ke * c - ko * sn);
  k[base + 1] = to_bf16(ke * sn + ko * c);
}

// ---------------------------------------------------------------------------
// Flash attention. Grid: x = S/128, y = B*H. Block = 8 waves; each wave owns
// 16 queries; 64-key tiles with online softmax. K/V tiles are staged in LDS
// (double buffered, async copies) and shared by all 8 waves of the block.
// Dynamic LDS layout (88,064 B):
//   [0)       K tiles: 2 x (64 rows x 256B @ pitch 272B) = 2 x 17408
//   [34816)   V tiles: 2 x (128 rows x 128B @ pitch 144B) = 2 x 18432
//   [71680)   P transpose buffers: 8 waves x 2048B
// ---------------------------------------------------------------------------
#define F_KPITCH 136          // elements (272 B)
#define F_VPITCH 72           // elements (144 B)
#define F_KBUF   17408u
#define F_VBUF   18432u
#define F_VOFF   34816u
#define F_POFF   71680u
#define F_LDS_SZ 88064u

__device__ __forceinline__ void flash_issue_tile(
    const bf16_t* __restrict__ Kc, const bf16_t* __restrict__ Vt,
    size_t bh, int key0, unsigned smem_base, int buf, int tid) {
  const unsigned kbase = smem_base + (unsigned)buf * F_KBUF;
  const unsigned vbase = smem_base + F_VOFF + (unsigned)buf * F_VBUF;
  const bf16_t* kg = Kc + (bh * 2048 + key0) * 128;
  const bf16_t* vg = Vt + bh * 128 * 2048 + key0;
#pragma unroll
  for (int u = 0; u < 4; ++u) {                 // K: 1024 16B chunks (64x16)
    const int idx = tid + 256 * u;
    const int row = idx >> 4, c = idx & 15;
    async_copy16(kg + (size_t)row * 128 + c * 8,
                 kbase + (unsigned)row * 272u + (unsigned)c * 16u);
  }
#pragma unroll
  for (int u = 0; u < 4; ++u) {                 // V: 1024 16B chunks (128x8)
    const int idx = tid + 256 * u;
    const int row = idx >> 3, c = idx & 7;
    async_copy16(vg + (size_t)row * 2048 + c * 8,
                 vbase + (unsigned)row * 144u + (unsigned)c * 16u);
  }
}

__global__ __launch_bounds__(256) void flash_attn_kernel(
    const bf16_t* __restrict__ Q, const bf16_t* __restrict__ Kc,
    const bf16_t* __restrict__ Vt, const float* __restrict__ mask,
    bf16_t* __restrict__ ctx) {
  extern __shared__ __align__(16) char fsm[];
  const int tid = threadIdx.x;
  const int lane = tid & 31;
  const int wave = tid >> 5;
  const int lrow = lane & 15, hi = lane >> 4;
  const int bhi = blockIdx.y;
  const int b = bhi >> 4, h = bhi & 15;
  const size_t bh = (size_t)bhi;
  const int m0 = blockIdx.x * 128 + wave * 16;
  const float scale = 0.08838834764831845f;   // 1/sqrt(128)
  const unsigned smem_base = lds_off_of(fsm);

  const bf16_t* qp = Q + (bh * 2048 + m0) * 128;
  bf16x16 qf[4];
#pragma unroll
  for (int kf = 0; kf < 4; ++kf) qf[kf] = load_frag_a(qp + 32 * kf, 128, lane);

  f32x8 o[8];
#pragma unroll
  for (int df = 0; df < 8; ++df) o[df] = (f32x8){};
  float mi[8], li[8];
#pragma unroll
  for (int r = 0; r < 8; ++r) { mi[r] = -1e30f; li[r] = 0.f; }

  flash_issue_tile(Kc, Vt, bh, 0, smem_base, 0, tid);

  for (int kt = 0; kt < 32; ++kt) {
    const int cur = kt & 1;
    const int key0 = kt << 6;
    wait_async0();
    __syncthreads();
    if (kt + 1 < 32)
      flash_issue_tile(Kc, Vt, bh, (kt + 1) << 6, smem_base, cur ^ 1, tid);

    const bf16_t* KL = (const bf16_t*)(fsm + cur * F_KBUF);
    const bf16_t* VL = (const bf16_t*)(fsm + F_VOFF + cur * F_VBUF);

    // ---- scores S = (Q @ K^T) * scale + mask -----------------------------
    f32x8 sf[4];
#pragma unroll
    for (int nf = 0; nf < 4; ++nf) {
      f32x8 a = {};
      const bf16_t* kp = KL + (16 * nf) * F_KPITCH;
#pragma unroll
      for (int kf = 0; kf < 4; ++kf)
        a = wmma_bf16(qf[kf], load_frag_b(kp + 32 * kf, F_KPITCH, lane), a);
      sf[nf] = a;
    }
    float rmax[8];
#pragma unroll
    for (int r = 0; r < 8; ++r) rmax[r] = -1e30f;
#pragma unroll
    for (int nf = 0; nf < 4; ++nf) {
      const float mv = mask[b * 2048 + key0 + 16 * nf + lrow];
#pragma unroll
      for (int r = 0; r < 8; ++r) {
        const float s = sf[nf][r] * scale + mv;
        sf[nf][r] = s;
        rmax[r] = fmaxf(rmax[r], s);
      }
    }
    // ---- online softmax stats (reduce across the 16-lane column group) ---
#pragma unroll
    for (int r = 0; r < 8; ++r)
#pragma unroll
      for (int off = 1; off < 16; off <<= 1)
        rmax[r] = fmaxf(rmax[r], __shfl_xor(rmax[r], off, 32));
    float corr[8], rsum[8];
#pragma unroll
    for (int r = 0; r < 8; ++r) {
      const float mnew = fmaxf(mi[r], rmax[r]);
      corr[r] = __expf(mi[r] - mnew);
      mi[r] = mnew;
      rsum[r] = 0.f;
    }
#pragma unroll
    for (int nf = 0; nf < 4; ++nf)
#pragma unroll
      for (int r = 0; r < 8; ++r) {
        const float p = __expf(sf[nf][r] - mi[r]);
        sf[nf][r] = p;
        rsum[r] += p;
      }
#pragma unroll
    for (int r = 0; r < 8; ++r) {
#pragma unroll
      for (int off = 1; off < 16; off <<= 1)
        rsum[r] += __shfl_xor(rsum[r], off, 32);
      li[r] = li[r] * corr[r] + rsum[r];
    }
#pragma unroll
    for (int df = 0; df < 8; ++df)
#pragma unroll
      for (int r = 0; r < 8; ++r) o[df][r] *= corr[r];

    // ---- transpose P (C layout) -> A layout via LDS ----------------------
    bf16_t* pb = (bf16_t*)(fsm + F_POFF) + wave * 1024;
#pragma unroll
    for (int nf = 0; nf < 4; ++nf)
#pragma unroll
      for (int r = 0; r < 8; ++r)
        pb[(r + 8 * hi) * 64 + nf * 16 + lrow] = to_bf16(sf[nf][r]);
    __syncthreads();

    bf16x16 pa[2];
#pragma unroll
    for (int k2 = 0; k2 < 2; ++k2) pa[k2] = load_frag_a(pb + 32 * k2, 64, lane);

    // ---- O += P @ V ------------------------------------------------------
#pragma unroll
    for (int df = 0; df < 8; ++df) {
      const bf16_t* vp = VL + (16 * df) * F_VPITCH;
#pragma unroll
      for (int k2 = 0; k2 < 2; ++k2)
        o[df] = wmma_bf16(pa[k2], load_frag_b(vp + 32 * k2, F_VPITCH, lane), o[df]);
    }
  }

  // ---- normalize and write ctx -------------------------------------------
#pragma unroll
  for (int df = 0; df < 8; ++df) {
    const int d = df * 16 + lrow;
#pragma unroll
    for (int r = 0; r < 8; ++r) {
      const int s = m0 + r + 8 * hi;
      const float val = o[df][r] / li[r];
      ctx[((size_t)(b * 2048 + s)) * 2048 + h * 128 + d] = to_bf16(val);
    }
  }
}

// ---------------------------------------------------------------------------
extern "C" void kernel_launch(void* const* d_in, const int* in_sizes, int n_in,
                              void* d_out, int out_size, void* d_ws, size_t ws_size,
                              hipStream_t stream) {
  const float* hs   = (const float*)d_in[0];   // [2,2048,2048]
  const float* mask = (const float*)d_in[1];   // [2,1,1,2048]
  const float* Wqkv = (const float*)d_in[2];   // [6144,2048]
  const float* bqkv = (const float*)d_in[3];   // [6144]
  const float* Wout = (const float*)d_in[4];   // [2048,2048]
  const float* bout = (const float*)d_in[5];   // [2048]
  float* out = (float*)d_out;                  // [2,2048,2048]

  char* ws = (char*)d_ws;
  const size_t SZ_HS   = (size_t)4096 * 2048 * 2;   // 16,777,216
  const size_t SZ_WQKV = (size_t)6144 * 2048 * 2;   // 25,165,824
  const size_t SZ_WOUT = (size_t)2048 * 2048 * 2;   //  8,388,608
  const size_t SZ_QKV1 = (size_t)2 * 16 * 2048 * 128 * 2; // 16,777,216 each

  bf16_t* hs_b   = (bf16_t*)(ws);
  bf16_t* wqkv_b = (bf16_t*)(ws + SZ_HS);
  bf16_t* wout_b = (bf16_t*)(ws + SZ_HS + SZ_WQKV);
  size_t  o3     = SZ_HS + SZ_WQKV + SZ_WOUT;
  bf16_t* qb     = (bf16_t*)(ws + o3);
  bf16_t* kb     = (bf16_t*)(ws + o3 + SZ_QKV1);
  bf16_t* vtb    = (bf16_t*)(ws + o3 + 2 * SZ_QKV1);
  bf16_t* ctxb   = (bf16_t*)(ws + o3 + 3 * SZ_QKV1);
  // total workspace use: 117,440,512 bytes

  // 1) fp32 -> bf16 staging
  cvt_f32_bf16_kernel<<<(8388608 + 255) / 256, 256, 0, stream>>>(hs, hs_b, 8388608);
  cvt_f32_bf16_kernel<<<(12582912 + 255) / 256, 256, 0, stream>>>(Wqkv, wqkv_b, 12582912);
  cvt_f32_bf16_kernel<<<(4194304 + 255) / 256, 256, 0, stream>>>(Wout, wout_b, 4194304);

  // 2) QKV projection + scatter (q,k -> [B,H,S,D]; v -> [B,H,D,S])
  gemm_bf16_kernel<<<dim3(6144 / 256, 4096 / 128), 256, 0, stream>>>(
      hs_b, wqkv_b, bqkv, nullptr, qb, kb, vtb, 4096, 6144, 2048, 1);

  // 3) RoPE in place on q,k
  rope_kernel<<<4194304 / 256, 256, 0, stream>>>(qb, kb);

  // 4) flash attention -> ctx bf16 [4096,2048]
  flash_attn_kernel<<<dim3(2048 / 128, 32), 256, F_LDS_SZ, stream>>>(
      qb, kb, vtb, mask, ctxb);

  // 5) output projection -> fp32 d_out
  gemm_bf16_kernel<<<dim3(2048 / 256, 4096 / 128), 256, 0, stream>>>(
      ctxb, wout_b, bout, out, nullptr, nullptr, nullptr, 4096, 2048, 2048, 0);
}
